// MPLayer_22746146799734
// MI455X (gfx1250) — matrix operand order repaired and checked by reading, not compiled, not run on hardware
//
#include <hip/hip_runtime.h>
#include <hip/hip_bf16.h>

#define IN_FEATS  128
#define OUT_FEATS 128

typedef __attribute__((ext_vector_type(16))) __bf16 v16bf;
typedef __attribute__((ext_vector_type(8)))  __bf16 v8bf;
typedef __attribute__((ext_vector_type(4)))  __bf16 v4bf;
typedef __attribute__((ext_vector_type(8)))  float  v8f;
typedef __attribute__((ext_vector_type(4)))  float  v4f;

// LDS row stride for the staged weight matrix: 128 bf16 + 8 pad = 272 bytes
// = 68 dwords -> each successive row rotates 4 banks; a wave32 ds_load_b128
// of a B fragment hits every LDS bank exactly twice (minimum possible).
#define WLDS_STRIDE 136

// -------------------------------------------------------------------------
// Zero a float buffer (grid-stride, float4 stores). Used for agg each call.
// -------------------------------------------------------------------------
__global__ __launch_bounds__(256) void zero_f32_kernel(float* __restrict__ p, int n4) {
    int i      = blockIdx.x * 256 + threadIdx.x;
    int stride = gridDim.x * 256;
    v4f z = {0.f, 0.f, 0.f, 0.f};
    for (; i < n4; i += stride) ((v4f*)p)[i] = z;
}

// -------------------------------------------------------------------------
// Y[m, n] = (relu?) ( sum_k X[m,k] * W[n,k] + bias[n] )
// X: [M,128] f32 row-major. W: [128,128] f32 row-major (row = out feature),
// converted to bf16 into LDS once per workgroup and shared by all 8 waves.
// One wave32 computes a 16(M) x 128(N) strip via 8 WMMA accumulator tiles.
//
// wave32 WMMA layouts (CDNA5 ISA 7.12.2):
//   A 16x32 bf16 : lane L holds row M=L&15; VGPR0-3 = K kbase..kbase+7,
//                  VGPR4-7 = K kbase+16..+23, kbase = (L>>4)*8  (per 32-K step)
//   B 32x16 bf16 : lane L holds col N=L&15; VGPR0-3 = K kbase..kbase+7,
//                  VGPR4-7 = K kbase+16..+23 (as A) -> two 16B LDS row reads
//   C/D 16x16 f32: VGPR i -> M = i + (L>>4)*8, N = L&15
// -------------------------------------------------------------------------
__global__ __launch_bounds__(256) void gemm_bf16_wmma_kernel(
    const float* __restrict__ X,
    const float* __restrict__ W,
    const float* __restrict__ bias,
    float*       __restrict__ Y,
    int M, int relu)
{
    __shared__ __bf16 ldsW[128 * WLDS_STRIDE];   // ~34 KB of the 320 KB WGP LDS

    const int tid = threadIdx.x;

    // ---- stage W into LDS as bf16 (coalesced b128 global reads) ----
#pragma unroll
    for (int e4 = tid; e4 < (128 * 128) / 4; e4 += 256) {
        const int e = e4 * 4;
        const int n = e >> 7;
        const int k = e & 127;
        v4f w = *(const v4f*)(W + e);
        v4bf c;
        c[0] = (__bf16)w[0]; c[1] = (__bf16)w[1];
        c[2] = (__bf16)w[2]; c[3] = (__bf16)w[3];
        *(v4bf*)(&ldsW[n * WLDS_STRIDE + k]) = c;   // 8B-aligned ds_store
    }
    __syncthreads();

    const int lane    = tid & 31;
    const int wave    = tid >> 5;
    const int rowBase = (blockIdx.x * 8 + wave) * 16;   // wave-uniform

    if (rowBase < M) {                                  // uniform per wave
        const int mSub   = lane & 15;
        const int hiHalf = lane >> 4;

        // A-row pointer (clamped per-lane for safe loads; stores are guarded)
        int mA = rowBase + mSub;
        if (mA >= M) mA = M - 1;
        const float* __restrict__ xrow = X + (size_t)mA * IN_FEATS;

        // accumulators initialized with bias[n] (n depends only on lane&15)
        v8f acc[8];
#pragma unroll
        for (int t = 0; t < 8; ++t) {
            float bv = bias[t * 16 + mSub];
            acc[t] = (v8f){bv, bv, bv, bv, bv, bv, bv, bv};
        }

        // prime the A pipeline for ks = 0
        v4f pr0, pr1, pr2, pr3;
        {
            const float* xk = xrow + hiHalf * 8;
            pr0 = *(const v4f*)(xk);
            pr1 = *(const v4f*)(xk + 4);
            pr2 = *(const v4f*)(xk + 16);
            pr3 = *(const v4f*)(xk + 20);
        }

        // K loop: 128 = 4 steps of 32, fully unrolled + software pipelined
#pragma unroll
        for (int ks = 0; ks < 4; ++ks) {
            v16bf a;
#pragma unroll
            for (int j = 0; j < 4; ++j) {
                a[j]      = (__bf16)pr0[j];   // K = kbase..kbase+3
                a[4 + j]  = (__bf16)pr1[j];   // K = kbase+4..kbase+7
                a[8 + j]  = (__bf16)pr2[j];   // K = kbase+16..kbase+19
                a[12 + j] = (__bf16)pr3[j];   // K = kbase+20..kbase+23
            }
            if (ks < 3) {   // issue next K-step's A loads 8 WMMAs early
                const float* xk = xrow + (ks + 1) * 32 + hiHalf * 8;
                pr0 = *(const v4f*)(xk);
                pr1 = *(const v4f*)(xk + 4);
                pr2 = *(const v4f*)(xk + 16);
                pr3 = *(const v4f*)(xk + 20);
            }

            const int kcol = ks * 32 + hiHalf * 8;
            // prime B pipeline: tile 0 fragment from LDS
            v8bf blo = *(const v8bf*)(&ldsW[mSub * WLDS_STRIDE + kcol]);
            v8bf bhi = *(const v8bf*)(&ldsW[mSub * WLDS_STRIDE + kcol + 16]);
#pragma unroll
            for (int t = 0; t < 8; ++t) {
                v16bf b = __builtin_shufflevector(
                    blo, bhi, 0, 1, 2, 3, 4, 5, 6, 7, 8, 9, 10, 11, 12, 13, 14, 15);
                if (t < 7) {  // issue next tile's LDS loads before this WMMA
                    const int nrow = (t + 1) * 16 + mSub;
                    blo = *(const v8bf*)(&ldsW[nrow * WLDS_STRIDE + kcol]);
                    bhi = *(const v8bf*)(&ldsW[nrow * WLDS_STRIDE + kcol + 16]);
                }
                acc[t] = __builtin_amdgcn_wmma_f32_16x16x32_bf16(
                             false, a, false, b, (short)0, acc[t], false, false);
            }
        }

        // store (fused ReLU), guarded for the ragged last tile
#pragma unroll
        for (int t = 0; t < 8; ++t) {
            const int n = t * 16 + mSub;
#pragma unroll
            for (int i = 0; i < 8; ++i) {
                const int m = rowBase + hiHalf * 8 + i;
                if (m < M) {
                    float v = acc[t][i];
                    if (relu) v = fmaxf(v, 0.0f);
                    Y[(size_t)m * OUT_FEATS + n] = v;
                }
            }
        }
    }
}

// -------------------------------------------------------------------------
// One wave32 per edge: lane l moves h[src][4l..4l+3] into agg[dst][...] via
// f32 atomics (resolve in L2 — both h and agg fit in the 192 MB L2).
// -------------------------------------------------------------------------
__global__ __launch_bounds__(256) void edge_scatter_kernel(
    const float* __restrict__ h,
    const int*   __restrict__ src,
    const int*   __restrict__ dst,
    float*       __restrict__ agg,
    int nEdges)
{
    const int gid  = blockIdx.x * 256 + threadIdx.x;
    const int edge = gid >> 5;
    const int lane = gid & 31;
    if (edge >= nEdges) return;

    const int s = src[edge];
    const int d = dst[edge];
    v4f v = *(const v4f*)(h + (size_t)s * IN_FEATS + lane * 4);
    float* out = agg + (size_t)d * IN_FEATS + lane * 4;
    atomicAdd(out + 0, v[0]);
    atomicAdd(out + 1, v[1]);
    atomicAdd(out + 2, v[2]);
    atomicAdd(out + 3, v[3]);
}

// -------------------------------------------------------------------------
extern "C" void kernel_launch(void* const* d_in, const int* in_sizes, int n_in,
                              void* d_out, int out_size, void* d_ws, size_t ws_size,
                              hipStream_t stream) {
    const float* node_feats = (const float*)d_in[0];
    const int*   src        = (const int*)  d_in[1];
    const int*   dst        = (const int*)  d_in[2];
    const float* W_msg      = (const float*)d_in[3];
    const float* b_msg      = (const float*)d_in[4];
    const float* W_out      = (const float*)d_in[5];
    const float* b_out      = (const float*)d_in[6];
    float* out = (float*)d_out;

    const int nNodes = in_sizes[0] / IN_FEATS;
    const int nEdges = in_sizes[1];

    // workspace: [ agg : nNodes*128 f32 ]
    float* agg = (float*)d_ws;

    // 1) zero agg (ws is not re-poisoned between replays; must not carry state)
    const int n4 = nNodes * IN_FEATS / 4;
    zero_f32_kernel<<<2048, 256, 0, stream>>>(agg, n4);

    // 2) h = relu(X @ W_msg.T + b_msg)  -> staged in d_out
    const int rowTiles   = (nNodes + 15) / 16;
    const int gemmBlocks = (rowTiles + 7) / 8;   // 8 waves/block, 1 tile/wave
    gemm_bf16_wmma_kernel<<<gemmBlocks, 256, 0, stream>>>(
        node_feats, W_msg, b_msg, out, nNodes, 1);

    // 3) agg[dst] += h[src]  (wave per edge)
    const long long sThreads = (long long)nEdges * 32;
    const int sBlocks = (int)((sThreads + 255) / 256);
    edge_scatter_kernel<<<sBlocks, 256, 0, stream>>>(out, src, dst, agg, nEdges);

    // 4) out = agg @ W_out.T + b_out  (overwrites the h staging buffer)
    gemm_bf16_wmma_kernel<<<gemmBlocks, 256, 0, stream>>>(
        agg, W_out, b_out, out, nNodes, 0);
}
